// Decoder_25726854103192
// MI455X (gfx1250) — matrix operand order repaired and checked by reading, not compiled, not run on hardware
//
#include <hip/hip_runtime.h>
#include <hip/hip_bf16.h>
#include <math.h>

typedef _Float16 htype;
typedef __attribute__((ext_vector_type(16))) _Float16 v16h;
typedef __attribute__((ext_vector_type(8)))  _Float16 v8h;
typedef __attribute__((ext_vector_type(8)))  float    v8f;

#define UNITS 512
#define VOCAB 32000
#define BATCH 64
#define TGT   32
#define SRC   32
#define KCONV 3
#define XCOLS (TGT + KCONV - 1)   // 34

// ---------------------------------------------------------------- helpers
__device__ __forceinline__ float gelu_f(float x) {
  const float c = 0.7978845608028654f;
  return 0.5f * x * (1.0f + tanhf(c * (x + 0.044715f * x * x * x)));
}
__device__ __forceinline__ float sigmoid_f(float x) { return 1.0f / (1.0f + __expf(-x)); }

// A tile (16x32 f16, M x K). Lane L holds row (L&15).
// lanes 0-15: elem e -> K = e (e<8), e+8 (e>=8)   => chunks at K offsets {0,16}
// lanes16-31: elem e -> K = e+8 (e<8), e+16 (e>=8)=> chunks at K offsets {8,24}
__device__ __forceinline__ v16h load_a_tile(const htype* base, int ld) {
  const int lane = threadIdx.x & 31;
  const htype* p = base + (lane & 15) * ld + ((lane < 16) ? 0 : 8);
  v8h lo = *(const v8h*)(p);
  v8h hi = *(const v8h*)(p + 16);
  v16h r;
#pragma unroll
  for (int i = 0; i < 8; ++i) { r[i] = lo[i]; r[i + 8] = hi[i]; }
  return r;
}
// B tile (32x16 f16, K x N) read from row-major [N][K] weight (y = x @ W^T).
// Lane L holds column n=(L&15); lanes 0-15: K={0..7,16..23}, lanes 16-31: K={8..15,24..31}.
__device__ __forceinline__ v16h load_b_tile(const htype* base, int ld) {
  const int lane = threadIdx.x & 31;
  const htype* p = base + (lane & 15) * ld + ((lane < 16) ? 0 : 8);
  v8h lo = *(const v8h*)(p);
  v8h hi = *(const v8h*)(p + 16);
  v16h r;
#pragma unroll
  for (int i = 0; i < 8; ++i) { r[i] = lo[i]; r[i + 8] = hi[i]; }
  return r;
}
__device__ __forceinline__ v8f wmma_f16(v16h a, v16h b, v8f c) {
  return __builtin_amdgcn_wmma_f32_16x16x32_f16(false, a, false, b, (short)0, c, false, false);
}

// ---------------------------------------------------------------- generic WMMA GEMM
// C[m,n] = sum_k A[m,k] * W[n,k]  (+ bias[n]) (+ Cin[m,n]) ; optional gelu; optional f16 copy out.
// One wave32 per 16x16 tile; dual accumulators break the WMMA->WMMA RAW chain.
// grid = (M/16, N/16), block = 32.  K must be a multiple of 64 (all uses: 512).
__global__ __launch_bounds__(32) void wmma_gemm_kernel(
    const htype* __restrict__ A, int lda, const htype* __restrict__ W, int ldw,
    const float* __restrict__ bias, const float* __restrict__ Cin, int ldci,
    float* __restrict__ Cout, int ldc, htype* __restrict__ CoutH, int ldch,
    int K, int do_gelu) {
  const int mt = blockIdx.x, nt = blockIdx.y;
  const htype* a_base = A + (size_t)mt * 16 * lda;
  const htype* w_base = W + (size_t)nt * 16 * ldw;
  v8f acc0 = {}, acc1 = {};
  for (int k = 0; k < K; k += 64) {
    acc0 = wmma_f16(load_a_tile(a_base + k, lda), load_b_tile(w_base + k, ldw), acc0);
    acc1 = wmma_f16(load_a_tile(a_base + k + 32, lda), load_b_tile(w_base + k + 32, ldw), acc1);
  }
  const int lane = threadIdx.x;
  const int n = nt * 16 + (lane & 15);
  const int m0 = mt * 16 + ((lane < 16) ? 0 : 8);
  const float bn = bias ? bias[n] : 0.0f;
#pragma unroll
  for (int r = 0; r < 8; ++r) {
    const int m = m0 + r;
    float v = acc0[r] + acc1[r] + bn;
    if (Cin) v += Cin[(size_t)m * ldci + n];
    if (do_gelu) v = gelu_f(v);
    Cout[(size_t)m * ldc + n] = v;
    if (CoutH) CoutH[(size_t)m * ldch + n] = (htype)v;
  }
}

// ---------------------------------------------------------------- conv-as-GEMM (3 taps)
// X: [B][34][512] f16.  Wt: [3][512][512] f16 with Wt[k][o][i] = conv_w[o][i][k].
// mode 0: gelu epilogue, store f16 to outH[b][t+2][o]   (feeds conv2's input)
// mode 1: + resid (embedded), store f32 to outF[m][o]
__global__ __launch_bounds__(32) void wmma_conv_kernel(
    const htype* __restrict__ X, const htype* __restrict__ Wt,
    const float* __restrict__ bias, const float* __restrict__ resid,
    float* __restrict__ outF, htype* __restrict__ outH, int mode) {
  const int mt = blockIdx.x;            // 128 tiles of 16 rows (m = b*32 + t)
  const int nt = blockIdx.y;            // 32 tiles of 16 output channels
  const int m0 = mt * 16;
  const int b = m0 >> 5;                // 32 rows per batch; tiles never cross b
  const int t0 = m0 & 31;
  const htype* xb = X + (size_t)b * XCOLS * UNITS;
  v8f acc0 = {}, acc1 = {};
#pragma unroll
  for (int kt = 0; kt < KCONV; ++kt) {
    const htype* a_base = xb + (t0 + kt) * UNITS;
    const htype* w_base = Wt + (size_t)kt * UNITS * UNITS + (size_t)nt * 16 * UNITS;
    for (int kk = 0; kk < UNITS; kk += 64) {
      acc0 = wmma_f16(load_a_tile(a_base + kk, UNITS), load_b_tile(w_base + kk, UNITS), acc0);
      acc1 = wmma_f16(load_a_tile(a_base + kk + 32, UNITS), load_b_tile(w_base + kk + 32, UNITS), acc1);
    }
  }
  const int lane = threadIdx.x;
  const int o = nt * 16 + (lane & 15);
  const int r0 = (lane < 16) ? 0 : 8;
  const float bn = bias[o];
#pragma unroll
  for (int r = 0; r < 8; ++r) {
    const int m = m0 + r0 + r;
    const int t = m & 31;
    float v = acc0[r] + acc1[r] + bn;
    if (mode == 0) {
      v = gelu_f(v);
      outH[(size_t)b * XCOLS * UNITS + (t + 2) * UNITS + o] = (htype)v;
    } else {
      v += resid[(size_t)m * UNITS + o];
      outF[(size_t)m * UNITS + o] = v;
    }
  }
}

// ---------------------------------------------------------------- big GEMM (logits)
// 64x64 output per wave: 4 M-tiles x 4 N-tiles, 16 independent accumulators.
// Per K-step: 4 A-tile loads + 4 B-tile loads feed 16 WMMAs (~0.5 KB moved per WMMA),
// and each accumulator's RAW distance is 16 WMMAs -> matrix pipe stays full.
__global__ __launch_bounds__(32) void wmma_gemm_big_kernel(
    const htype* __restrict__ A, const htype* __restrict__ W,
    const float* __restrict__ bias, float* __restrict__ C, int N, int K) {
  const int mt0 = blockIdx.x * 4;
  const int nt0 = blockIdx.y * 4;
  const htype* a0 = A + (size_t)mt0 * 16 * K;
  const htype* w0 = W + (size_t)nt0 * 16 * K;
  __builtin_prefetch(a0, 0, 1);                 // global_prefetch_b8
  __builtin_prefetch(w0, 0, 1);
  v8f acc[4][4] = {};
  for (int k = 0; k < K; k += 32) {
    v16h av[4];
#pragma unroll
    for (int i = 0; i < 4; ++i)
      av[i] = load_a_tile(a0 + (size_t)i * 16 * K + k, K);
#pragma unroll
    for (int j = 0; j < 4; ++j) {
      v16h bv = load_b_tile(w0 + (size_t)j * 16 * K + k, K);
#pragma unroll
      for (int i = 0; i < 4; ++i)
        acc[i][j] = wmma_f16(av[i], bv, acc[i][j]);
    }
  }
  const int lane = threadIdx.x;
  const int nloc = lane & 15;
  const int rbase = (lane < 16) ? 0 : 8;
#pragma unroll
  for (int j = 0; j < 4; ++j) {
    const int n = (nt0 + j) * 16 + nloc;
    const float bn = bias[n];
#pragma unroll
    for (int i = 0; i < 4; ++i) {
      const int mrow = (mt0 + i) * 16 + rbase;
#pragma unroll
      for (int r = 0; r < 8; ++r)
        C[(size_t)(mrow + r) * N + n] = acc[i][j][r] + bn;
    }
  }
}

// ---------------------------------------------------------------- layernorm (row len 512)
__global__ void layernorm_kernel(const float* __restrict__ x, int ldx,
                                 const float* __restrict__ y, int ldy,
                                 const float* __restrict__ g, const float* __restrict__ be,
                                 float* __restrict__ of, int ldo,
                                 htype* __restrict__ oh, int ldoh) {
  const int row = blockIdx.x;
  const float* xr = x + (size_t)row * ldx;
  const float* yr = y ? y + (size_t)row * ldy : nullptr;
  __shared__ float red[256];
  float s = 0.0f;
  for (int u = threadIdx.x; u < UNITS; u += 256) s += xr[u] + (yr ? yr[u] : 0.0f);
  red[threadIdx.x] = s; __syncthreads();
  for (int off = 128; off > 0; off >>= 1) { if (threadIdx.x < off) red[threadIdx.x] += red[threadIdx.x + off]; __syncthreads(); }
  const float mu = red[0] / (float)UNITS; __syncthreads();
  float s2 = 0.0f;
  for (int u = threadIdx.x; u < UNITS; u += 256) {
    float v = xr[u] + (yr ? yr[u] : 0.0f) - mu; s2 += v * v;
  }
  red[threadIdx.x] = s2; __syncthreads();
  for (int off = 128; off > 0; off >>= 1) { if (threadIdx.x < off) red[threadIdx.x] += red[threadIdx.x + off]; __syncthreads(); }
  const float rstd = rsqrtf(red[0] / (float)UNITS + 1e-5f); __syncthreads();
  for (int u = threadIdx.x; u < UNITS; u += 256) {
    float v = (xr[u] + (yr ? yr[u] : 0.0f) - mu) * rstd * g[u] + be[u];
    of[(size_t)row * ldo + u] = v;
    if (oh) oh[(size_t)row * ldoh + u] = (htype)v;
  }
}

// ---------------------------------------------------------------- GRU gates (biases already in gi/gh)
__global__ void gru_gates_kernel(const float* __restrict__ gi, const float* __restrict__ gh,
                                 float* __restrict__ h, htype* __restrict__ hh) {
  const int i = blockIdx.x * 256 + threadIdx.x;
  if (i >= BATCH * UNITS) return;
  const int b = i >> 9, u = i & 511;
  const size_t gb = (size_t)b * 3 * UNITS;
  const float i_r = gi[gb + u], i_z = gi[gb + UNITS + u], i_n = gi[gb + 2 * UNITS + u];
  const float h_r = gh[gb + u], h_z = gh[gb + UNITS + u], h_n = gh[gb + 2 * UNITS + u];
  const float r = sigmoid_f(i_r + h_r);
  const float z = sigmoid_f(i_z + h_z);
  const float n = tanhf(i_n + r * h_n);
  const float hp = h[i];
  const float hn = (1.0f - z) * n + z * hp;
  h[i] = hn;
  hh[i] = (htype)hn;
}

// ---------------------------------------------------------------- attention (one block per batch row)
__global__ void attention_kernel(const float* __restrict__ q, const float* __restrict__ ctx,
                                 const unsigned char* __restrict__ mask,
                                 float* __restrict__ aout, float* __restrict__ attn_out) {
  const int b = blockIdx.x, tid = threadIdx.x;
  __shared__ float sc[SRC], sw[SRC], ms[2];
  if (tid < SRC) {
    const float* qb = q + (size_t)b * UNITS;
    const float* cb = ctx + ((size_t)b * SRC + tid) * UNITS;
    float dot = 0.0f;
    for (int u = 0; u < UNITS; ++u) dot += qb[u] * cb[u];
    dot *= 0.044194173824159216f;                 // 1/sqrt(512)
    if (mask[b * SRC + tid]) dot = -1.0e9f;
    sc[tid] = dot;
  }
  __syncthreads();
  if (tid == 0) {
    float m = -1.0e30f;
    for (int s = 0; s < SRC; ++s) m = fmaxf(m, sc[s]);
    float sum = 0.0f;
    for (int s = 0; s < SRC; ++s) sum += __expf(sc[s] - m);
    ms[0] = m; ms[1] = sum;
  }
  __syncthreads();
  if (tid < SRC) {
    const float w = __expf(sc[tid] - ms[0]) / ms[1];
    sw[tid] = w;
    attn_out[(size_t)b * TGT * SRC + tid] = w;    // caller offsets by t*SRC
  }
  __syncthreads();
  for (int u = tid; u < UNITS; u += blockDim.x) {
    float acc = 0.0f;
    for (int s = 0; s < SRC; ++s) acc += sw[s] * ctx[((size_t)b * SRC + s) * UNITS + u];
    aout[(size_t)b * UNITS + u] = acc;
  }
}

// ---------------------------------------------------------------- vocab softmax (in place, row 32000)
__global__ void softmax_kernel(float* __restrict__ x, int N) {
  const int row = blockIdx.x, tid = threadIdx.x;
  float* xr = x + (size_t)row * N;
  __shared__ float red[256];
  float m = -1.0e30f;
  for (int i = tid; i < N; i += 256) m = fmaxf(m, xr[i]);
  red[tid] = m; __syncthreads();
  for (int off = 128; off > 0; off >>= 1) { if (tid < off) red[tid] = fmaxf(red[tid], red[tid + off]); __syncthreads(); }
  const float rm = red[0]; __syncthreads();
  float s = 0.0f;
  for (int i = tid; i < N; i += 256) s += __expf(xr[i] - rm);
  red[tid] = s; __syncthreads();
  for (int off = 128; off > 0; off >>= 1) { if (tid < off) red[tid] += red[tid + off]; __syncthreads(); }
  const float inv = 1.0f / red[0]; __syncthreads();
  for (int i = tid; i < N; i += 256) xr[i] = __expf(xr[i] - rm) * inv;
}

// ---------------------------------------------------------------- small utility kernels
__global__ void f32_to_f16_kernel(const float* __restrict__ in, htype* __restrict__ out, int n) {
  const int i = blockIdx.x * 256 + threadIdx.x;
  if (i < n) out[i] = (htype)in[i];
}
// conv weight repack: (O,I,K) f32 -> [k][o][i] f16
__global__ void convw_kernel(const float* __restrict__ w, htype* __restrict__ out, int n) {
  const int i = blockIdx.x * 256 + threadIdx.x;
  if (i >= n) return;
  const int k = i >> 18;          // /(512*512)
  const int rem = i & 262143;
  const int o = rem >> 9, ii = rem & 511;
  out[i] = (htype)w[(size_t)o * (UNITS * KCONV) + ii * KCONV + k];
}
// Build x (and the mem-prefix of h): xh[b][s][u]; s<2 from cnn_mem (b,u,s); else embedding
__global__ void build_x_kernel(const int* __restrict__ seq, const float* __restrict__ emb,
                               const float* __restrict__ mem0, const float* __restrict__ mem1,
                               htype* __restrict__ xh, htype* __restrict__ hh,
                               float* __restrict__ embedded) {
  const int i = blockIdx.x * 256 + threadIdx.x;
  if (i >= BATCH * XCOLS * UNITS) return;
  const int u = i & 511;
  const int s = (i >> 9) % XCOLS;
  const int b = i / (XCOLS * UNITS);
  if (s < 2) {
    xh[i] = (htype)mem0[(size_t)b * UNITS * 2 + u * 2 + s];
    hh[i] = (htype)mem1[(size_t)b * UNITS * 2 + u * 2 + s];
  } else {
    const int t = s - 2;
    const int tok = seq[b * TGT + t];
    const float e = emb[(size_t)tok * UNITS + u];
    xh[i] = (htype)e;
    embedded[((size_t)b * TGT + t) * UNITS + u] = e;
  }
}
__global__ void gather_last_kernel(const float* __restrict__ outs, float* __restrict__ dst) {
  const int i = blockIdx.x * 256 + threadIdx.x;
  if (i >= BATCH * UNITS) return;
  const int b = i >> 9, u = i & 511;
  dst[i] = outs[((size_t)b * TGT + (TGT - 1)) * UNITS + u];
}

// ================================================================ host
extern "C" void kernel_launch(void* const* d_in, const int* in_sizes, int n_in,
                              void* d_out, int out_size, void* d_ws, size_t ws_size,
                              hipStream_t stream) {
  (void)in_sizes; (void)n_in; (void)out_size; (void)ws_size;
  const int*   seq      = (const int*)d_in[0];
  const float* context  = (const float*)d_in[1];
  const float* state    = (const float*)d_in[2];
  const float* fh_in    = (const float*)d_in[3];
  const unsigned char* pad_mask = (const unsigned char*)d_in[4];
  const float* cnn_mem0 = (const float*)d_in[5];
  const float* cnn_mem1 = (const float*)d_in[6];
  const float* emb      = (const float*)d_in[7];
  const float* conv1_w  = (const float*)d_in[8];
  const float* conv1_b  = (const float*)d_in[9];
  const float* conv2_w  = (const float*)d_in[10];
  const float* conv2_b  = (const float*)d_in[11];
  const float* w_ih     = (const float*)d_in[12];
  const float* w_hh     = (const float*)d_in[13];
  const float* b_ih     = (const float*)d_in[14];
  const float* b_hh     = (const float*)d_in[15];
  const float* g1 = (const float*)d_in[16]; const float* be1 = (const float*)d_in[17];
  const float* g2 = (const float*)d_in[18]; const float* be2 = (const float*)d_in[19];
  const float* g3 = (const float*)d_in[20]; const float* be3 = (const float*)d_in[21];
  const float* wq     = (const float*)d_in[22];
  const float* fc1_w  = (const float*)d_in[23];
  const float* fc1_b  = (const float*)d_in[24];
  const float* fc2_w  = (const float*)d_in[25];
  const float* fc2_b  = (const float*)d_in[26];

  // d_out regions (floats)
  float* prob    = (float*)d_out;                                  // 64*32*32000
  float* h_fin   = prob + (size_t)BATCH * TGT * VOCAB;             // 64*512
  float* fh_fin  = h_fin + BATCH * UNITS;                          // 64*512
  float* o_mem0  = fh_fin + BATCH * UNITS;                         // 64*512*2
  float* o_mem1  = o_mem0 + BATCH * UNITS * 2;                     // 64*512*2
  float* attn    = o_mem1 + BATCH * UNITS * 2;                     // 64*32*32

  // workspace bump allocator
  char* wsb = (char*)d_ws;
  size_t off = 0;
  auto alloc = [&](size_t bytes) -> char* {
    char* p = wsb + off;
    off += (bytes + 255) & ~(size_t)255;
    return p;
  };
  htype* w1h   = (htype*)alloc((size_t)KCONV * UNITS * UNITS * 2);
  htype* w2h   = (htype*)alloc((size_t)KCONV * UNITS * UNITS * 2);
  htype* wihh  = (htype*)alloc((size_t)3 * UNITS * 2 * UNITS * 2);
  htype* whhh  = (htype*)alloc((size_t)3 * UNITS * UNITS * 2);
  htype* wqh   = (htype*)alloc((size_t)UNITS * UNITS * 2);
  htype* fc1h  = (htype*)alloc((size_t)UNITS * UNITS * 2);
  htype* fc2h  = (htype*)alloc((size_t)VOCAB * UNITS * 2);
  htype* xh    = (htype*)alloc((size_t)BATCH * XCOLS * UNITS * 2);
  htype* hhx   = (htype*)alloc((size_t)BATCH * XCOLS * UNITS * 2);
  float* embedded = (float*)alloc((size_t)BATCH * TGT * UNITS * 4);
  float* cnnsum   = (float*)alloc((size_t)BATCH * TGT * UNITS * 4);
  float* cnnoutf  = (float*)alloc((size_t)BATCH * TGT * UNITS * 4);
  htype* cnnouth  = (htype*)alloc((size_t)BATCH * TGT * UNITS * 2);
  float* gibuf    = (float*)alloc((size_t)BATCH * 3 * UNITS * 4);
  float* ghbuf    = (float*)alloc((size_t)BATCH * 3 * UNITS * 4);
  float* hf       = (float*)alloc((size_t)BATCH * UNITS * 4);
  htype* hsth     = (htype*)alloc((size_t)BATCH * UNITS * 2);
  htype* fh0h     = (htype*)alloc((size_t)BATCH * UNITS * 2);
  float* routf    = (float*)alloc((size_t)BATCH * UNITS * 4);
  htype* routh    = (htype*)alloc((size_t)BATCH * UNITS * 2);
  float* qf       = (float*)alloc((size_t)BATCH * UNITS * 4);
  float* af       = (float*)alloc((size_t)BATCH * UNITS * 4);
  float* af2      = (float*)alloc((size_t)BATCH * UNITS * 4);
  htype* ah       = (htype*)alloc((size_t)BATCH * UNITS * 2);
  float* outsf    = (float*)alloc((size_t)BATCH * TGT * UNITS * 4);
  htype* outsh    = (htype*)alloc((size_t)BATCH * TGT * UNITS * 2);

  const int TU = TGT * UNITS;   // 16384 row stride of (B,T,U) tensors

  // ---- weight conversion / repack
  convw_kernel<<<(KCONV * UNITS * UNITS + 255) / 256, 256, 0, stream>>>(conv1_w, w1h, KCONV * UNITS * UNITS);
  convw_kernel<<<(KCONV * UNITS * UNITS + 255) / 256, 256, 0, stream>>>(conv2_w, w2h, KCONV * UNITS * UNITS);
  f32_to_f16_kernel<<<(3 * UNITS * 2 * UNITS + 255) / 256, 256, 0, stream>>>(w_ih, wihh, 3 * UNITS * 2 * UNITS);
  f32_to_f16_kernel<<<(3 * UNITS * UNITS + 255) / 256, 256, 0, stream>>>(w_hh, whhh, 3 * UNITS * UNITS);
  f32_to_f16_kernel<<<(UNITS * UNITS + 255) / 256, 256, 0, stream>>>(wq, wqh, UNITS * UNITS);
  f32_to_f16_kernel<<<(UNITS * UNITS + 255) / 256, 256, 0, stream>>>(fc1_w, fc1h, UNITS * UNITS);
  f32_to_f16_kernel<<<(VOCAB * UNITS + 255) / 256, 256, 0, stream>>>(fc2_w, fc2h, VOCAB * UNITS);
  f32_to_f16_kernel<<<(BATCH * UNITS + 255) / 256, 256, 0, stream>>>(state, hsth, BATCH * UNITS);
  f32_to_f16_kernel<<<(BATCH * UNITS + 255) / 256, 256, 0, stream>>>(fh_in, fh0h, BATCH * UNITS);
  hipMemcpyAsync(hf, state, (size_t)BATCH * UNITS * 4, hipMemcpyDeviceToDevice, stream);

  // ---- embedding + CNN front end
  build_x_kernel<<<(BATCH * XCOLS * UNITS + 255) / 256, 256, 0, stream>>>(seq, emb, cnn_mem0, cnn_mem1, xh, hhx, embedded);
  wmma_conv_kernel<<<dim3(128, 32), 32, 0, stream>>>(xh, w1h, conv1_b, nullptr, nullptr, hhx, 0);
  wmma_conv_kernel<<<dim3(128, 32), 32, 0, stream>>>(hhx, w2h, conv2_b, embedded, cnnsum, nullptr, 1);
  layernorm_kernel<<<BATCH * TGT, 256, 0, stream>>>(cnnsum, UNITS, nullptr, 0, g1, be1, cnnoutf, UNITS, cnnouth, UNITS);

  // ---- recurrence over 32 timesteps
  for (int t = 0; t < TGT; ++t) {
    const htype* fhA = (t == 0) ? fh0h : (outsh + (size_t)(t - 1) * UNITS);
    const int fhlda = (t == 0) ? UNITS : TU;
    // gi = c_t @ w_ih[:, :512]^T + b_ih
    wmma_gemm_kernel<<<dim3(4, 96), 32, 0, stream>>>(
        cnnouth + (size_t)t * UNITS, TU, wihh, 2 * UNITS, b_ih, nullptr, 0,
        gibuf, 3 * UNITS, nullptr, 0, UNITS, 0);
    // gi += fh_prev @ w_ih[:, 512:]^T
    wmma_gemm_kernel<<<dim3(4, 96), 32, 0, stream>>>(
        fhA, fhlda, wihh + UNITS, 2 * UNITS, nullptr, gibuf, 3 * UNITS,
        gibuf, 3 * UNITS, nullptr, 0, UNITS, 0);
    // gh = h_prev @ w_hh^T + b_hh
    wmma_gemm_kernel<<<dim3(4, 96), 32, 0, stream>>>(
        hsth, UNITS, whhh, UNITS, b_hh, nullptr, 0,
        ghbuf, 3 * UNITS, nullptr, 0, UNITS, 0);
    // gates -> h (in place) + f16 copy
    gru_gates_kernel<<<(BATCH * UNITS + 255) / 256, 256, 0, stream>>>(gibuf, ghbuf, hf, hsth);
    // r_out = LN(c_t + h_new)
    layernorm_kernel<<<BATCH, 256, 0, stream>>>(cnnoutf + (size_t)t * UNITS, TU, hf, UNITS,
                                                g2, be2, routf, UNITS, routh, UNITS);
    // q = r_out @ wq^T
    wmma_gemm_kernel<<<dim3(4, 32), 32, 0, stream>>>(
        routh, UNITS, wqh, UNITS, nullptr, nullptr, 0, qf, UNITS, nullptr, 0, UNITS, 0);
    // attention + weights out
    attention_kernel<<<BATCH, 256, 0, stream>>>(qf, context, pad_mask, af, attn + (size_t)t * SRC);
    // a = LN(a + r_out)
    layernorm_kernel<<<BATCH, 256, 0, stream>>>(af, UNITS, routf, UNITS, g3, be3, af2, UNITS, ah, UNITS);
    // out = gelu(a @ fc1_w^T + fc1_b) -> outs[:, t, :] (f32 + f16)
    wmma_gemm_kernel<<<dim3(4, 32), 32, 0, stream>>>(
        ah, UNITS, fc1h, UNITS, fc1_b, nullptr, 0,
        outsf + (size_t)t * UNITS, TU, outsh + (size_t)t * UNITS, TU, UNITS, 1);
  }

  // ---- final projection (2048 x 32000 x 512) + softmax, logits straight into d_out
  wmma_gemm_big_kernel<<<dim3(BATCH * TGT / 64, VOCAB / 64), 32, 0, stream>>>(outsh, fc2h, fc2_b, prob, VOCAB, UNITS);
  softmax_kernel<<<BATCH * TGT, 256, 0, stream>>>(prob, VOCAB);

  // ---- remaining outputs
  hipMemcpyAsync(h_fin, hf, (size_t)BATCH * UNITS * 4, hipMemcpyDeviceToDevice, stream);
  gather_last_kernel<<<(BATCH * UNITS + 255) / 256, 256, 0, stream>>>(outsf, fh_fin);
  hipMemcpyAsync(o_mem0, cnn_mem0, (size_t)BATCH * UNITS * 2 * 4, hipMemcpyDeviceToDevice, stream);
  hipMemcpyAsync(o_mem1, cnn_mem1, (size_t)BATCH * UNITS * 2 * 4, hipMemcpyDeviceToDevice, stream);
}
